// SmyrfEncoderBlock_62886911148419
// MI455X (gfx1250) — matrix operand order repaired and checked by reading, not compiled, not run on hardware
//
#include <hip/hip_runtime.h>
#include <math.h>

typedef __attribute__((ext_vector_type(16))) _Float16 v16h;
typedef __attribute__((ext_vector_type(8)))  _Float16 v8h;
typedef __attribute__((ext_vector_type(8)))  float    v8f;

namespace {

constexpr int kB = 4, kL = 2048, kE = 768, kH = 12, kMLP = 3072, kDH = 64;
constexpr int kNH = 4;
constexpr int kBS = kB * kH;          // 48
constexpr int kRows = kB * kL;        // 8192
constexpr int kNPos = kBS * kL;       // 98304
constexpr int kNC = kL / 128;         // 16 clusters per sequence

__device__ __forceinline__ v16h pack16(v8h lo, v8h hi) {
  v16h r;
#pragma unroll
  for (int i = 0; i < 8; ++i) { r[i] = lo[i]; r[8 + i] = hi[i]; }
  return r;
}

__device__ __forceinline__ v8f wmma16(v16h a, v16h b, v8f c) {
  // D = A(16x32 f16) * B(32x16 f16) + C(16x16 f32)
  return __builtin_amdgcn_wmma_f32_16x16x32_f16(false, a, false, b, (short)0, c,
                                                false, false);
}

__device__ __forceinline__ v8f vzero8() {
  v8f z = {0.f, 0.f, 0.f, 0.f, 0.f, 0.f, 0.f, 0.f};
  return z;
}

// low 32 bits of a flat pointer to __shared__ == LDS byte offset (ISA 10.2)
__device__ __forceinline__ unsigned lds_off(const void* p) {
  return (unsigned)(unsigned long long)p;
}

// async DMA 16B from global -> LDS, tracked by ASYNCcnt (gfx1250)
__device__ __forceinline__ void async_copy16(unsigned lds_byte_off,
                                             const void* gptr) {
  asm volatile("global_load_async_to_lds_b128 %0, %1, off"
               :
               : "v"(lds_byte_off), "v"(gptr)
               : "memory");
}

__device__ __forceinline__ void wait_async0() {
  asm volatile("s_wait_asynccnt 0x0" ::: "memory");
}

// ---------------- elementwise helpers ----------------
__global__ void cast_f16_kernel(const float* __restrict__ in,
                                _Float16* __restrict__ out, long n) {
  long i = (long)blockIdx.x * blockDim.x + threadIdx.x;
  long stride = (long)gridDim.x * blockDim.x;
  for (; i < n; i += stride) out[i] = (_Float16)in[i];
}

__global__ void zero2_kernel(int* p) {
  if (threadIdx.x < 2) p[threadIdx.x] = 0;
}

// ---------------- layernorm -> f16 ----------------
__global__ __launch_bounds__(256) void ln_kernel(const float* __restrict__ x,
                                                 const float* __restrict__ g,
                                                 const float* __restrict__ b,
                                                 _Float16* __restrict__ out) {
  int row = blockIdx.x;
  const float* xr = x + (long)row * kE;
  int tid = threadIdx.x, lane = tid & 31, w = tid >> 5;
  float s = 0.f, s2 = 0.f;
  for (int i = tid; i < kE; i += 256) { float v = xr[i]; s += v; s2 += v * v; }
#pragma unroll
  for (int m = 16; m > 0; m >>= 1) {
    s += __shfl_xor(s, m, 32);
    s2 += __shfl_xor(s2, m, 32);
  }
  __shared__ float rs[8], rs2[8];
  if (lane == 0) { rs[w] = s; rs2[w] = s2; }
  __syncthreads();
  s = 0.f; s2 = 0.f;
#pragma unroll
  for (int i = 0; i < 8; ++i) { s += rs[i]; s2 += rs2[i]; }
  float mean = s * (1.f / kE);
  float var = s2 * (1.f / kE) - mean * mean;
  float inv = rsqrtf(var + 1e-6f);
  _Float16* orow = out + (long)row * kE;
  for (int i = tid; i < kE; i += 256)
    orow[i] = (_Float16)((xr[i] - mean) * inv * g[i] + b[i]);
}

// ---------------- WMMA GEMM: C[m,n] = sum_k A[m,k]*W[n,k] + bias[n] ------
// Double-buffered: async global->LDS DMA of K-slice p^1 overlaps WMMA on p.
enum { EPI_QKV = 0, EPI_BIAS_RES = 1, EPI_GELU = 2 };

template <int MODE>
__global__ __launch_bounds__(256) void gemm_kernel(
    const _Float16* __restrict__ A, const _Float16* __restrict__ W,
    const float* __restrict__ bias, int N, int K,
    float* __restrict__ out_f32, const float* __restrict__ res,
    _Float16* __restrict__ out_f16,
    _Float16* __restrict__ qo, _Float16* __restrict__ ko,
    _Float16* __restrict__ vo, float* __restrict__ qo32,
    float* __restrict__ ko32) {
  constexpr int LD = 40;  // 32 halves + 8 pad (80B rows, 16B aligned)
  __shared__ _Float16 As[2][128 * LD];
  __shared__ _Float16 Bs[2][128 * LD];
  const int bn0 = blockIdx.x * 128, bm0 = blockIdx.y * 128;
  const int tid = threadIdx.x, lane = tid & 31, w = tid >> 5;
  const int wr = (w >> 2) * 64, wc = (w & 3) * 32;  // wave 64x32 sub-tile
  const int g = lane >> 4, ln = lane & 15;
  v8f acc[4][2];
#pragma unroll
  for (int i = 0; i < 4; ++i)
#pragma unroll
    for (int j = 0; j < 2; ++j) acc[i][j] = vzero8();

  // this thread's 16B staging slots: row r, half-row cc
  const int r = tid >> 1, cc = (tid & 1) * 16;
  const _Float16* ga = A + (long)(bm0 + r) * K + cc;
  const _Float16* gb = W + (long)(bn0 + r) * K + cc;
  unsigned la[2], lb[2];
#pragma unroll
  for (int p = 0; p < 2; ++p) {
    la[p] = lds_off(&As[p][r * LD + cc]);
    lb[p] = lds_off(&Bs[p][r * LD + cc]);
  }

  auto issue = [&](int p, int k0) {
    async_copy16(la[p], ga + k0);
    async_copy16(la[p] + 16, ga + k0 + 8);
    async_copy16(lb[p], gb + k0);
    async_copy16(lb[p] + 16, gb + k0 + 8);
  };
  auto compute = [&](int p) {
    v16h af[4], bf[2];
#pragma unroll
    for (int i = 0; i < 4; ++i) {
      const _Float16* ap = &As[p][(wr + i * 16 + ln) * LD];
      af[i] = pack16(*(const v8h*)(ap + g * 16), *(const v8h*)(ap + g * 16 + 8));
    }
#pragma unroll
    for (int j = 0; j < 2; ++j) {
      const _Float16* bp = &Bs[p][(wc + j * 16 + ln) * LD + g * 16];
      bf[j] = pack16(*(const v8h*)(bp), *(const v8h*)(bp + 8));
    }
#pragma unroll
    for (int i = 0; i < 4; ++i)
#pragma unroll
      for (int j = 0; j < 2; ++j) acc[i][j] = wmma16(af[i], bf[j], acc[i][j]);
  };

  issue(0, 0);
  wait_async0();
  __syncthreads();
  int p = 0;
  for (int k0 = 32; k0 < K; k0 += 32) {
    issue(p ^ 1, k0);   // DMA next slice while matrix pipe works on slice p
    compute(p);
    wait_async0();
    __syncthreads();    // ds reads of buf p done; buf p^1 tile visible
    p ^= 1;
  }
  compute(p);

  // epilogue: C layout -> lane (g,ln): row = 8*g+vv within tile, col = ln
#pragma unroll
  for (int i = 0; i < 4; ++i)
#pragma unroll
    for (int j = 0; j < 2; ++j)
#pragma unroll
      for (int vv = 0; vv < 8; ++vv) {
        int m = bm0 + wr + i * 16 + 8 * g + vv;
        int n = bn0 + wc + j * 16 + ln;
        float val = acc[i][j][vv] + bias[n];
        if (MODE == EPI_QKV) {
          int which = n / kE, col = n - which * kE;
          int hh = col >> 6, d = col & 63;
          int bb = m >> 11, l = m & 2047;
          long di = (((long)(bb * kH + hh)) * kL + l) * kDH + d;
          if (which == 0) {
            float sv = val * 0.125f;  // DH^-0.5
            qo[di] = (_Float16)sv;
            qo32[di] = sv;
          } else if (which == 1) {
            ko[di] = (_Float16)val;
            ko32[di] = val;
          } else {
            vo[di] = (_Float16)val;
          }
        } else if (MODE == EPI_BIAS_RES) {
          long idx = (long)m * N + n;
          out_f32[idx] = val + res[idx];
        } else {  // EPI_GELU (exact erf)
          float ge = 0.5f * val * (1.f + erff(val * 0.70710678118654752f));
          out_f16[(long)m * N + n] = (_Float16)ge;
        }
      }
}

// ---------------- XBOX+ norms + global max ----------------
__global__ __launch_bounds__(256) void sqmax_kernel(
    const float* __restrict__ q32, const float* __restrict__ k32,
    float* __restrict__ qsq, float* __restrict__ ksq, int* __restrict__ mx) {
  int row = blockIdx.x * 256 + threadIdx.x;  // [0, kNPos)
  const float* qr = q32 + (long)row * kDH;
  const float* kr = k32 + (long)row * kDH;
  float a = 0.f, c = 0.f;
#pragma unroll 8
  for (int d = 0; d < kDH; ++d) {
    float qv = qr[d], kv = kr[d];
    a += qv * qv;
    c += kv * kv;
  }
  qsq[row] = a;
  ksq[row] = c;
  float am = a, cm = c;
#pragma unroll
  for (int m = 16; m > 0; m >>= 1) {
    am = fmaxf(am, __shfl_xor(am, m, 32));
    cm = fmaxf(cm, __shfl_xor(cm, m, 32));
  }
  if ((threadIdx.x & 31) == 0) {
    atomicMax(&mx[0], __float_as_int(am));  // values >= 0: bits monotonic
    atomicMax(&mx[1], __float_as_int(cm));
  }
}

// ---------------- E2LSH projections ----------------
__global__ __launch_bounds__(256) void keys_kernel(
    const float* __restrict__ q32, const float* __restrict__ k32,
    const float* __restrict__ qsq, const float* __restrict__ ksq,
    const int* __restrict__ mx, const float* __restrict__ alpha,
    const float* __restrict__ beta, float* __restrict__ qkeys,
    float* __restrict__ kkeys) {
  int row = blockIdx.x * 256 + threadIdx.x;
  float mtot = __int_as_float(mx[0]) + __int_as_float(mx[1]);
  float qa = sqrtf(fmaxf(mtot - qsq[row], 0.f));
  float ka = sqrtf(fmaxf(mtot - ksq[row], 0.f));
  const float* qr = q32 + (long)row * kDH;
  const float* kr = k32 + (long)row * kDH;
  float aq[kNH], ak[kNH];
#pragma unroll
  for (int h = 0; h < kNH; ++h) { aq[h] = beta[h]; ak[h] = beta[h]; }
  for (int d = 0; d < kDH; ++d) {
    float qv = qr[d], kv = kr[d];
#pragma unroll
    for (int h = 0; h < kNH; ++h) {
      float al = alpha[d * kNH + h];
      aq[h] += qv * al;
      ak[h] += kv * al;
    }
  }
#pragma unroll
  for (int h = 0; h < kNH; ++h) {
    aq[h] += qa * alpha[kDH * kNH + h];          // Qx aug at index DH
    ak[h] += ka * alpha[(kDH + 1) * kNH + h];    // Kx aug at index DH+1
    qkeys[(long)h * kNPos + row] = aq[h];
    kkeys[(long)h * kNPos + row] = ak[h];
  }
}

// ---------------- per-(hash,head) bitonic argsort over L=2048 -----------
__global__ __launch_bounds__(256) void sort_kernel(const float* __restrict__ keys,
                                                   int* __restrict__ pos) {
  __shared__ float kb[kL];
  __shared__ int ib[kL];
  int row = blockIdx.x;  // hash*48 + head
  const float* src = keys + (long)row * kL;
  for (int i = threadIdx.x; i < kL; i += 256) { kb[i] = src[i]; ib[i] = i; }
  __syncthreads();
  for (int sz = 2; sz <= kL; sz <<= 1) {
    for (int st = sz >> 1; st > 0; st >>= 1) {
      for (int i = threadIdx.x; i < kL; i += 256) {
        int j = i ^ st;
        if (j > i) {
          bool up = ((i & sz) == 0);
          float ki = kb[i], kj = kb[j];
          if (up ? (ki > kj) : (ki < kj)) {
            kb[i] = kj; kb[j] = ki;
            int t = ib[i]; ib[i] = ib[j]; ib[j] = t;
          }
        }
      }
      __syncthreads();
    }
  }
  int* dst = pos + (long)row * kL;
  for (int i = threadIdx.x; i < kL; i += 256) dst[i] = ib[i];
}

// ---------------- clustered attention (one block per hash*head*cluster) --
__global__ __launch_bounds__(256) void smyrf_attn_kernel(
    const _Float16* __restrict__ q, const _Float16* __restrict__ k,
    const _Float16* __restrict__ v, const int* __restrict__ q_pos,
    const int* __restrict__ k_pos, _Float16* __restrict__ bo,
    float* __restrict__ lse_out) {
  constexpr int LDV = 136, LDP = 136;  // padded rows, 16B-aligned
  __shared__ _Float16 Vt[64 * LDV];    // V transposed: [d][kv]
  __shared__ _Float16 Ps[128 * LDP];   // probabilities [q][kv]
  const int blk = blockIdx.x;
  const int c = blk & (kNC - 1);
  const int hb = blk >> 4;  // hash*48 + head
  const int bsi = hb % kBS;
  const long qkb = (long)bsi * kL * kDH;
  const int posbase = hb * kL + c * 128;
  const int tid = threadIdx.x;
  // stage V transposed via k_pos gather
  for (int e = tid; e < 128 * 64; e += 256) {
    int kk = e >> 6, d = e & 63;
    int krow = k_pos[posbase + kk];
    Vt[d * LDV + kk] = v[qkb + (long)krow * kDH + d];
  }
  __syncthreads();
  const int lane = tid & 31, w = tid >> 5;
  const int g = lane >> 4, ln = lane & 15;
  const int qr = q_pos[posbase + w * 16 + ln];
  const _Float16* qrow = q + qkb + (long)qr * kDH;
  int krj[8];
#pragma unroll
  for (int j = 0; j < 8; ++j) krj[j] = k_pos[posbase + j * 16 + ln];
  // S = q . k  (wave w owns q-rows [w*16, w*16+16), all 128 kv columns)
  v8f acc[8];
#pragma unroll
  for (int j = 0; j < 8; ++j) acc[j] = vzero8();
#pragma unroll
  for (int kk = 0; kk < 64; kk += 32) {
    v16h af = pack16(*(const v8h*)(qrow + kk + g * 16),
                     *(const v8h*)(qrow + kk + g * 16 + 8));
#pragma unroll
    for (int j = 0; j < 8; ++j) {
      const _Float16* krow = k + qkb + (long)krj[j] * kDH + kk + g * 16;
      v16h bfr = pack16(*(const v8h*)krow, *(const v8h*)(krow + 8));
      acc[j] = wmma16(af, bfr, acc[j]);
    }
  }
  // in-register row softmax (16-lane half-group shuffles)
  float lse_r[8];
#pragma unroll
  for (int vv = 0; vv < 8; ++vv) {
    float m = -3.4e38f;
#pragma unroll
    for (int j = 0; j < 8; ++j) m = fmaxf(m, acc[j][vv]);
#pragma unroll
    for (int msk = 1; msk < 16; msk <<= 1) m = fmaxf(m, __shfl_xor(m, msk, 32));
    float s = 0.f;
#pragma unroll
    for (int j = 0; j < 8; ++j) s += expf(acc[j][vv] - m);
#pragma unroll
    for (int msk = 1; msk < 16; msk <<= 1) s += __shfl_xor(s, msk, 32);
    float inv = 1.f / s;
    lse_r[vv] = m + logf(s);
    int rl = w * 16 + 8 * g + vv;
#pragma unroll
    for (int j = 0; j < 8; ++j)
      Ps[rl * LDP + j * 16 + ln] = (_Float16)(expf(acc[j][vv] - m) * inv);
  }
  __syncthreads();
  // bo = P @ V
  v8f acc2[4];
#pragma unroll
  for (int j = 0; j < 4; ++j) acc2[j] = vzero8();
  const _Float16* prow = Ps + (w * 16 + ln) * LDP;
#pragma unroll
  for (int kk = 0; kk < 128; kk += 32) {
    v16h af = pack16(*(const v8h*)(prow + kk + g * 16),
                     *(const v8h*)(prow + kk + g * 16 + 8));
#pragma unroll
    for (int j = 0; j < 4; ++j) {
      const _Float16* vrow = Vt + (j * 16 + ln) * LDV + kk + g * 16;
      v16h bfr = pack16(*(const v8h*)vrow, *(const v8h*)(vrow + 8));
      acc2[j] = wmma16(af, bfr, acc2[j]);
    }
  }
  // scatter by q_pos = inverse of the sort
#pragma unroll
  for (int vv = 0; vv < 8; ++vv) {
    int qi = w * 16 + 8 * g + vv;
    int orow = q_pos[posbase + qi];
    long ob = ((long)hb * kL + orow) * kDH;
#pragma unroll
    for (int j = 0; j < 4; ++j) bo[ob + j * 16 + ln] = (_Float16)acc2[j][vv];
    if (ln == 0) lse_out[(long)hb * kL + orow] = lse_r[vv];
  }
}

// ---------------- merge hash rounds, heads -> (B*L, E) f16 ----------------
__global__ void merge_kernel(const _Float16* __restrict__ bo,
                             const float* __restrict__ lse,
                             _Float16* __restrict__ attn) {
  long i = (long)blockIdx.x * 256 + threadIdx.x;
  if (i >= (long)kRows * kE) return;
  int col = (int)(i % kE);
  long m = i / kE;
  int bb = (int)(m >> 11), l = (int)(m & 2047);
  int hh = col >> 6, d = col & 63;
  int bsi = bb * kH + hh;
  float lg[kNH];
  float mx = -3.4e38f;
#pragma unroll
  for (int h = 0; h < kNH; ++h) {
    lg[h] = lse[((long)(h * kBS + bsi)) * kL + l];
    mx = fmaxf(mx, lg[h]);
  }
  float z = 0.f, o = 0.f;
#pragma unroll
  for (int h = 0; h < kNH; ++h) {
    float wgt = expf(lg[h] - mx);
    z += wgt;
    o += wgt * (float)bo[(((long)(h * kBS + bsi)) * kL + l) * kDH + d];
  }
  attn[i] = (_Float16)(o / z);
}

}  // namespace

extern "C" void kernel_launch(void* const* d_in, const int* in_sizes, int n_in,
                              void* d_out, int out_size, void* d_ws,
                              size_t ws_size, hipStream_t stream) {
  (void)in_sizes; (void)n_in; (void)out_size; (void)ws_size;
  const float* x     = (const float*)d_in[0];
  const float* ln1g  = (const float*)d_in[1];
  const float* ln1b  = (const float*)d_in[2];
  const float* wqkv  = (const float*)d_in[3];
  const float* bqkv  = (const float*)d_in[4];
  const float* wout  = (const float*)d_in[5];
  const float* bout  = (const float*)d_in[6];
  const float* ln2g  = (const float*)d_in[7];
  const float* ln2b  = (const float*)d_in[8];
  const float* w1    = (const float*)d_in[9];
  const float* b1    = (const float*)d_in[10];
  const float* w2    = (const float*)d_in[11];
  const float* b2    = (const float*)d_in[12];
  const float* alpha = (const float*)d_in[13];
  const float* beta  = (const float*)d_in[14];
  float* out = (float*)d_out;

  char* ws = (char*)d_ws;
  size_t off = 0;
  auto take = [&](size_t bytes) -> void* {
    void* p = ws + off;
    off += (bytes + 255) & ~(size_t)255;
    return p;
  };
  _Float16* wqkv16 = (_Float16*)take((size_t)3 * kE * kE * 2);
  _Float16* wout16 = (_Float16*)take((size_t)kE * kE * 2);
  _Float16* w116   = (_Float16*)take((size_t)kMLP * kE * 2);
  _Float16* w216   = (_Float16*)take((size_t)kE * kMLP * 2);
  _Float16* h16    = (_Float16*)take((size_t)kRows * kE * 2);  // reused as LN2 out
  _Float16* q16    = (_Float16*)take((size_t)kNPos * kDH * 2);
  _Float16* k16    = (_Float16*)take((size_t)kNPos * kDH * 2);
  _Float16* v16    = (_Float16*)take((size_t)kNPos * kDH * 2);
  float* q32       = (float*)take((size_t)kNPos * kDH * 4);
  float* k32       = (float*)take((size_t)kNPos * kDH * 4);
  float* qsq       = (float*)take((size_t)kNPos * 4);
  float* ksq       = (float*)take((size_t)kNPos * 4);
  int* mxbuf       = (int*)take(256);
  float* qkeys     = (float*)take((size_t)kNH * kNPos * 4);
  float* kkeys     = (float*)take((size_t)kNH * kNPos * 4);
  int* qpos        = (int*)take((size_t)kNH * kNPos * 4);
  int* kpos        = (int*)take((size_t)kNH * kNPos * 4);
  float* lsebuf    = (float*)take((size_t)kNH * kNPos * 4);
  _Float16* bo16   = (_Float16*)take((size_t)kNH * kNPos * kDH * 2);
  _Float16* attn16 = (_Float16*)take((size_t)kRows * kE * 2);
  float* x2        = (float*)take((size_t)kRows * kE * 4);
  _Float16* y1 = bo16;  // alias: bo consumed by merge before MLP1 writes y1

  zero2_kernel<<<1, 32, 0, stream>>>(mxbuf);
  cast_f16_kernel<<<1024, 256, 0, stream>>>(wqkv, wqkv16, (long)3 * kE * kE);
  cast_f16_kernel<<<1024, 256, 0, stream>>>(wout, wout16, (long)kE * kE);
  cast_f16_kernel<<<1024, 256, 0, stream>>>(w1, w116, (long)kMLP * kE);
  cast_f16_kernel<<<1024, 256, 0, stream>>>(w2, w216, (long)kE * kMLP);

  ln_kernel<<<kRows, 256, 0, stream>>>(x, ln1g, ln1b, h16);

  gemm_kernel<EPI_QKV><<<dim3(3 * kE / 128, kRows / 128), 256, 0, stream>>>(
      h16, wqkv16, bqkv, 3 * kE, kE, nullptr, nullptr, nullptr, q16, k16, v16,
      q32, k32);

  sqmax_kernel<<<kNPos / 256, 256, 0, stream>>>(q32, k32, qsq, ksq, mxbuf);
  keys_kernel<<<kNPos / 256, 256, 0, stream>>>(q32, k32, qsq, ksq, mxbuf, alpha,
                                               beta, qkeys, kkeys);
  sort_kernel<<<kNH * kBS, 256, 0, stream>>>(qkeys, qpos);
  sort_kernel<<<kNH * kBS, 256, 0, stream>>>(kkeys, kpos);

  smyrf_attn_kernel<<<kNH * kBS * kNC, 256, 0, stream>>>(q16, k16, v16, qpos,
                                                         kpos, bo16, lsebuf);
  merge_kernel<<<(unsigned)(((long)kRows * kE + 255) / 256), 256, 0, stream>>>(
      bo16, lsebuf, attn16);

  gemm_kernel<EPI_BIAS_RES><<<dim3(kE / 128, kRows / 128), 256, 0, stream>>>(
      attn16, wout16, bout, kE, kE, x2, x, nullptr, nullptr, nullptr, nullptr,
      nullptr, nullptr);

  ln_kernel<<<kRows, 256, 0, stream>>>(x2, ln2g, ln2b, h16);

  gemm_kernel<EPI_GELU><<<dim3(kMLP / 128, kRows / 128), 256, 0, stream>>>(
      h16, w116, b1, kMLP, kE, nullptr, nullptr, y1, nullptr, nullptr, nullptr,
      nullptr, nullptr);

  gemm_kernel<EPI_BIAS_RES><<<dim3(kE / 128, kRows / 128), 256, 0, stream>>>(
      y1, w216, b2, kE, kMLP, out, x2, nullptr, nullptr, nullptr, nullptr,
      nullptr, nullptr);
}
